// CapsuleLayer_1778116461334
// MI455X (gfx1250) — matrix operand order repaired and checked by reading, not compiled, not run on hardware
//
#include <hip/hip_runtime.h>

// Problem dims (fixed by reference)
#define A_DIM 64
#define B_DIM 128
#define D_DIM 256
#define H_DIM 64
#define C_DIM 32

typedef __bf16 v16bf __attribute__((ext_vector_type(16)));
typedef float  v8f   __attribute__((ext_vector_type(8)));

#define PSTRIDE 260      // prior-tile LDS row stride (fp32), padded
#define KC 32            // K-chunk per WMMA step
#define WT_STRIDE 40     // transposed-W LDS row stride (bf16), KC + 8 pad

// One workgroup = one (weightIdx, aIdx) routing problem.
// Computes P = X (M x 256) * W (256 x 256) with split-bf16 WMMA (fp32-grade),
// keeps P in LDS, runs 3 dynamic-routing iterations, writes the 256-wide
// routed vector to Out[aIdx*outStrideA + wIdx*outStrideW .. +255].
template<int M>
__global__ __launch_bounds__(256) void caps_stage(
    const float* __restrict__ Xbase,   // [nA][M][256]
    const float* __restrict__ Wbase,   // [nW][256][256]
    float* __restrict__ Out,
    int nA, int outStrideA, int outStrideW)
{
  extern __shared__ __align__(16) char smem[];
  // Prior tile (written AFTER the K loop) overlays the streaming chunk buffers.
  float*  P   = (float*)smem;                  // [M][PSTRIDE]
  __bf16* Xh  = (__bf16*)smem;                 // [M][KC]  hi parts
  __bf16* Xl  = Xh + M * KC;                   // [M][KC]  lo parts
  __bf16* Wth = Xl + M * KC;                   // [256][WT_STRIDE] (transposed)
  __bf16* Wtl = Wth + D_DIM * WT_STRIDE;
  float* tail   = (float*)(smem + (size_t)M * PSTRIDE * 4);
  float* logits = tail;                        // [M]
  float* probs  = tail + M;                    // [M]
  float* outvec = tail + 2 * M;                // [256]
  float* red    = tail + 2 * M + 256;          // [256]

  const int tid  = threadIdx.x;
  const int lane = tid & 31;
  const int wv   = tid >> 5;      // wave id 0..7
  const int lr   = lane & 15;
  const int hh   = lane >> 4;     // lane half (0/1)

  const int aIdx = blockIdx.x % nA;
  const int wIdx = blockIdx.x / nA;
  const float* Xg = Xbase + (size_t)aIdx * (M * D_DIM);
  const float* Wg = Wbase + (size_t)wIdx * (D_DIM * D_DIM);

  constexpr int MT = M / 16;      // M tiles (8 or 4)
  v8f acc[MT][2];
  #pragma unroll
  for (int m = 0; m < MT; ++m)
    #pragma unroll
    for (int j = 0; j < 2; ++j)
      acc[m][j] = (v8f){0.f,0.f,0.f,0.f,0.f,0.f,0.f,0.f};

  // ---- GEMM: stream K in chunks of 32 through LDS, split-bf16 x 3 WMMA ----
  for (int kc = 0; kc < D_DIM / KC; ++kc) {
    __syncthreads();   // previous iteration's fragment reads done
    // X chunk: M x 32 fp32 -> hi/lo bf16, row-major
    constexpr int XE = M * KC / 256;
    #pragma unroll
    for (int i = 0; i < XE; ++i) {
      int idx = i * 256 + tid;
      int r = idx >> 5, c = idx & 31;
      float v = Xg[r * D_DIM + kc * KC + c];
      __bf16 hi = (__bf16)v;
      Xh[r * KC + c] = hi;
      Xl[r * KC + c] = (__bf16)(v - (float)hi);
    }
    // W chunk: 32 x 256 fp32 -> hi/lo bf16, transposed (padded) for B frags
    #pragma unroll 4
    for (int r = 0; r < KC; ++r) {
      float v = Wg[(kc * KC + r) * D_DIM + tid];
      __bf16 hi = (__bf16)v;
      Wth[tid * WT_STRIDE + r] = hi;
      Wtl[tid * WT_STRIDE + r] = (__bf16)(v - (float)hi);
    }
    __syncthreads();

    // B fragments for this wave's two N tiles (col=lane&15, contiguous K=hh*16+e, same as A)
    v16bf bh[2], bl[2];
    #pragma unroll
    for (int j = 0; j < 2; ++j) {
      int col  = (wv * 2 + j) * 16 + lr;
      int base = col * WT_STRIDE + hh * 16;
      #pragma unroll
      for (int e = 0; e < 16; ++e) { bh[j][e] = Wth[base + e]; bl[j][e] = Wtl[base + e]; }
    }
    #pragma unroll
    for (int m = 0; m < MT; ++m) {
      // A fragment (row=lane&15, K=hh*16+e)
      int xb = (m * 16 + lr) * KC + hh * 16;
      v16bf ah, al;
      #pragma unroll
      for (int e = 0; e < 8; ++e) {
        ah[e] = Xh[xb + e];  ah[8 + e] = Xh[xb + 8 + e];
        al[e] = Xl[xb + e];  al[8 + e] = Xl[xb + 8 + e];
      }
      #pragma unroll
      for (int j = 0; j < 2; ++j) {
        acc[m][j] = __builtin_amdgcn_wmma_f32_16x16x32_bf16(false, ah, false, bh[j], (short)0, acc[m][j], false, false);
        acc[m][j] = __builtin_amdgcn_wmma_f32_16x16x32_bf16(false, ah, false, bl[j], (short)0, acc[m][j], false, false);
        acc[m][j] = __builtin_amdgcn_wmma_f32_16x16x32_bf16(false, al, false, bh[j], (short)0, acc[m][j], false, false);
      }
    }
  }
  __syncthreads();   // all chunk reads done before overlaying P

  // Spill accumulators to LDS prior tile (ISA C layout: row=m*16+hh*8+v, col=n*16+lr)
  #pragma unroll
  for (int m = 0; m < MT; ++m)
    #pragma unroll
    for (int j = 0; j < 2; ++j) {
      int col = (wv * 2 + j) * 16 + lr;
      #pragma unroll
      for (int v = 0; v < 8; ++v)
        P[(m * 16 + hh * 8 + v) * PSTRIDE + col] = acc[m][j][v];
    }
  if (tid < M) logits[tid] = 0.f;
  __syncthreads();

  // ---- dynamic routing, 3 iterations, all in LDS ----
  float lastOut = 0.f;
  for (int it = 0; it < 3; ++it) {
    // softmax over R=M
    float lv = (tid < M) ? logits[tid] : -3.0e38f;
    red[tid] = lv; __syncthreads();
    #pragma unroll
    for (int s = 128; s > 0; s >>= 1) { if (tid < s) red[tid] = fmaxf(red[tid], red[tid + s]); __syncthreads(); }
    float mx = red[0]; __syncthreads();
    float ev = (tid < M) ? __expf(lv - mx) : 0.f;
    red[tid] = ev; __syncthreads();
    #pragma unroll
    for (int s = 128; s > 0; s >>= 1) { if (tid < s) red[tid] += red[tid + s]; __syncthreads(); }
    float esum = red[0]; __syncthreads();
    if (tid < M) probs[tid] = ev / esum;
    __syncthreads();

    // s[d] = sum_b probs[b] * P[b][d]   (lane-consecutive columns: conflict-free)
    float sAcc = 0.f;
    for (int b = 0; b < M; ++b) sAcc += probs[b] * P[b * PSTRIDE + tid];
    red[tid] = sAcc * sAcc; __syncthreads();
    #pragma unroll
    for (int s = 128; s > 0; s >>= 1) { if (tid < s) red[tid] += red[tid + s]; __syncthreads(); }
    float sq = red[0]; __syncthreads();
    float ov = sAcc * (sq / (1.f + sq)) * rsqrtf(sq);   // squash
    outvec[tid] = ov;
    lastOut = ov;
    __syncthreads();

    if (it != 2) {
      // logits[b] += P[b] . out  — each wave handles M/8 rows, wave32 reduce
      const int RW = M / 8;
      for (int r = 0; r < RW; ++r) {
        int b = wv * RW + r;
        float p = 0.f;
        #pragma unroll
        for (int i = 0; i < 8; ++i) { int d = lane + i * 32; p += P[b * PSTRIDE + d] * outvec[d]; }
        #pragma unroll
        for (int off = 16; off > 0; off >>= 1) p += __shfl_down(p, off);
        if (lane == 0) logits[b] += p;
      }
      __syncthreads();
    }
  }

  Out[(size_t)aIdx * outStrideA + (size_t)wIdx * outStrideW + tid] = lastOut;
}

extern "C" void kernel_launch(void* const* d_in, const int* in_sizes, int n_in,
                              void* d_out, int out_size, void* d_ws, size_t ws_size,
                              hipStream_t stream) {
  const float* x  = (const float*)d_in[0];   // [A][B][D] fp32
  const float* w1 = (const float*)d_in[1];   // [H][D][D] fp32
  const float* w2 = (const float*)d_in[2];   // [C][D][D] fp32
  float* out  = (float*)d_out;               // [C][A][D] fp32
  float* out1 = (float*)d_ws;                // scratch: [A][H][D] = 4 MiB

  size_t smem1 = (size_t)(128 * PSTRIDE + 2 * 128 + 512) * 4;  // ~136 KB
  size_t smem2 = (size_t)( 64 * PSTRIDE + 2 *  64 + 512) * 4;  // ~69 KB

  // Stage 1: 64x64 routing problems over R=B=128; out1[a][h][:]
  caps_stage<128><<<dim3(H_DIM * A_DIM), dim3(256), smem1, stream>>>(
      x, w1, out1, A_DIM, H_DIM * D_DIM /*a stride*/, D_DIM /*h stride*/);

  // Stage 2: 32x64 routing problems over R=H=64; out[c][a][:]
  caps_stage<64><<<dim3(C_DIM * A_DIM), dim3(256), smem2, stream>>>(
      out1, w2, out, A_DIM, D_DIM /*a stride*/, A_DIM * D_DIM /*c stride*/);
}
